// ESCMLayer_16131897163839
// MI455X (gfx1250) — matrix operand order repaired and checked by reading, not compiled, not run on hardware
//
#include <hip/hip_runtime.h>
#include <hip/hip_bf16.h>

// ---------------------------------------------------------------------------
// Problem constants (from reference)
// ---------------------------------------------------------------------------
constexpr int BATCH_   = 16384;
constexpr int EMBD_    = 128;
constexpr int NFLD_    = 23;
constexpr int SEQ_     = 3;
constexpr int FEAT_    = NFLD_ * EMBD_;      // 2944
constexpr int NEXP_    = 8;
constexpr int ESZ_     = 1024;
constexpr int NTOT_    = NEXP_ * ESZ_;       // 8192
constexpr int TSZ_     = 512;
constexpr int NG_      = 3;

typedef __attribute__((ext_vector_type(16))) __bf16 v16bf;
typedef __attribute__((ext_vector_type(8)))  float  v8f;
typedef __attribute__((ext_vector_type(4)))  int    v4i;

union FragU { v16bf bf; v4i i[2]; };

// ---------------------------------------------------------------------------
// K0a: repack expert_w f32 [e][f][h] -> bf16 [f][e*1024+h]  (K-major B matrix)
// ---------------------------------------------------------------------------
__global__ __launch_bounds__(256) void repack_expert_w_kernel(
    const float* __restrict__ w, __hip_bfloat16* __restrict__ wq)
{
    size_t tid = (size_t)blockIdx.x * blockDim.x + threadIdx.x;
    size_t total = (size_t)NEXP_ * FEAT_ * ESZ_;
    if (tid >= total) return;
    int e = (int)(tid / ((size_t)FEAT_ * ESZ_));
    int r = (int)(tid % ((size_t)FEAT_ * ESZ_));
    int f = r / ESZ_;
    int h = r % ESZ_;
    wq[(size_t)f * NTOT_ + e * ESZ_ + h] = __float2bfloat16(w[tid]);
}

// ---------------------------------------------------------------------------
// K0b: cast tower_w f32 [g][h][t] -> bf16 same layout (already K x N row-major)
// ---------------------------------------------------------------------------
__global__ __launch_bounds__(256) void cast_bf16_kernel(
    const float* __restrict__ in, __hip_bfloat16* __restrict__ out, int n)
{
    int tid = blockIdx.x * blockDim.x + threadIdx.x;
    if (tid < n) out[tid] = __float2bfloat16(in[tid]);
}

// ---------------------------------------------------------------------------
// K1: embedding gather + seq-sum pooling -> bf16 concat [b][field*128+d]
// ---------------------------------------------------------------------------
__global__ __launch_bounds__(128) void embed_pool_kernel(
    const int*   __restrict__ idx,       // [NFLD][BATCH][SEQ]
    const float* __restrict__ emb,       // [NUM_EMB][128]
    __hip_bfloat16* __restrict__ concat) // [BATCH][FEAT]
{
    int b     = blockIdx.x;
    int field = blockIdx.y;
    int d     = threadIdx.x;
    const int* ip = idx + ((size_t)field * BATCH_ + b) * SEQ_;
    float s = 0.f;
    #pragma unroll
    for (int t = 0; t < SEQ_; ++t) {
        int e = ip[t];
        s += emb[(size_t)e * EMBD_ + d];
    }
    concat[(size_t)b * FEAT_ + field * EMBD_ + d] = __float2bfloat16(s);
}

// ---------------------------------------------------------------------------
// Async global->LDS helpers (CDNA5: GLOBAL_LOAD_ASYNC_TO_LDS_B128, ASYNCcnt)
// LDS byte address = low 32 bits of the generic pointer (flat->LDS truncation).
// ---------------------------------------------------------------------------
__device__ __forceinline__ void async_copy_b128(unsigned lds_addr, const void* gptr)
{
    asm volatile("global_load_async_to_lds_b128 %0, %1, off"
                 :: "v"(lds_addr), "v"(gptr) : "memory");
}
__device__ __forceinline__ void wait_asynccnt0()
{
    asm volatile("s_wait_asynccnt 0x0" ::: "memory");
}

// ---------------------------------------------------------------------------
// K2: tiled bf16 GEMM with WMMA, C = act(A*B + bias)
// A: MxK bf16 row-major; B: KxN bf16 row-major.
// Block tile 128x256, 8 waves, each wave owns a 64x64 tile (16 WMMAs/k-step).
// Double-buffered LDS, filled with async global->LDS b128 copies; per-thread
// global source pointers are strength-reduced to constant-stride bumps.
// ---------------------------------------------------------------------------
constexpr int TM = 128, TN = 256, TK = 32;
constexpr int LDA_B = 80;    // 64 bytes data + 16 pad per A row
constexpr int LDB_B = 528;   // 512 bytes data + 16 pad per B row
constexpr int ASZ = TM * LDA_B;   // 10240
constexpr int BSZ = TK * LDB_B;   // 16896

template<bool RELU, bool OUT_BF16>
__global__ __launch_bounds__(256) void gemm_bf16_kernel(
    const __hip_bfloat16* __restrict__ A,
    const __hip_bfloat16* __restrict__ Bm,
    const float* __restrict__ bias,
    void* __restrict__ C,
    int M, int N, int K)
{
    __shared__ unsigned char lsA[2 * ASZ];
    __shared__ unsigned char lsB[2 * BSZ];

    const int tid  = threadIdx.x;
    const int wave = tid >> 5;
    const int lane = tid & 31;
    const int nBase = blockIdx.x * TN;
    const int mBase = blockIdx.y * TM;
    const int wm = wave >> 2;            // 0..1 -> 64-row stripe
    const int wn = wave & 3;             // 0..3 -> 64-col stripe
    const int laneMod  = lane & 15;
    const int laneHalf = lane >> 4;

    const unsigned ldsA0 = (unsigned)(uintptr_t)&lsA[0];
    const unsigned ldsB0 = (unsigned)(uintptr_t)&lsB[0];

    // Per-thread staging sources (tile 0) + LDS slots; advance by constant
    // strides each k-step so the hot loop does only pointer bumps.
    const char* gA[2]; unsigned lA[2];
    #pragma unroll
    for (int it = 0; it < 2; ++it) {               // A: 512 x 16B chunks
        int c = tid + it * 256;
        int row = c >> 2, ch = c & 3;
        gA[it] = (const char*)A + ((size_t)(mBase + row) * K + ch * 8) * 2;
        lA[it] = ldsA0 + row * LDA_B + ch * 16;
    }
    const char* gB[4]; unsigned lB[4];
    #pragma unroll
    for (int it = 0; it < 4; ++it) {               // B: 1024 x 16B chunks
        int c = tid + it * 256;
        int row = c >> 5, ch = c & 31;
        gB[it] = (const char*)Bm + ((size_t)row * N + nBase + ch * 8) * 2;
        lB[it] = ldsB0 + row * LDB_B + ch * 16;
    }
    const size_t aStep = (size_t)TK * 2;           // 64 B along K
    const size_t bStep = (size_t)TK * N * 2;       // 32 rows of B

    auto stage = [&](int buf) {
        #pragma unroll
        for (int it = 0; it < 2; ++it) {
            async_copy_b128(lA[it] + buf * ASZ, gA[it]);
            gA[it] += aStep;
        }
        #pragma unroll
        for (int it = 0; it < 4; ++it) {
            async_copy_b128(lB[it] + buf * BSZ, gB[it]);
            gB[it] += bStep;
        }
    };

    v8f acc[4][4];
    #pragma unroll
    for (int i = 0; i < 4; ++i)
        #pragma unroll
        for (int j = 0; j < 4; ++j)
            #pragma unroll
            for (int r = 0; r < 8; ++r) acc[i][j][r] = 0.f;

    stage(0);
    int buf = 0;
    for (int k0 = 0; k0 < K; k0 += TK) {
        wait_asynccnt0();          // my async copies for `buf` complete
        __syncthreads();           // everyone's complete; prior reads retired
        if (k0 + TK < K) stage(buf ^ 1);

        const unsigned char* baseA = lsA + buf * ASZ;
        const unsigned char* baseB = lsB + buf * BSZ;

        // B fragments: lane holds column n = laneMod; elem s -> k=8*half+s (s<8), 16+8*half+(s-8)
        FragU bfr[4];
        #pragma unroll
        for (int j = 0; j < 4; ++j) {
            const unsigned char* p = baseB + (wn * 64 + j * 16 + laneMod) * 2;
            #pragma unroll
            for (int s = 0; s < 16; ++s) {
                const int kk = (s < 8) ? (8 * laneHalf + s) : (16 + 8 * laneHalf + s - 8);
                bfr[j].bf[s] = *(const __bf16*)(p + kk * LDB_B);
            }
        }
        // A fragments: lane holds row m; VGPR 0-3 -> k=8*half+{0..7}, 4-7 -> +16
        FragU afr[4];
        #pragma unroll
        for (int i = 0; i < 4; ++i) {
            const unsigned char* p = baseA + (wm * 64 + i * 16 + laneMod) * LDA_B
                                   + laneHalf * 16;
            afr[i].i[0] = *(const v4i*)(p);
            afr[i].i[1] = *(const v4i*)(p + 32);
        }
        // 16 WMMAs per k-step per wave
        #pragma unroll
        for (int i = 0; i < 4; ++i)
            #pragma unroll
            for (int j = 0; j < 4; ++j)
                acc[i][j] = __builtin_amdgcn_wmma_f32_16x16x32_bf16(
                    false, afr[i].bf, false, bfr[j].bf,
                    (short)0, acc[i][j], false, false);
        buf ^= 1;
    }

    // epilogue: C layout VGPR r -> M = r (lanes 0-15) / 8+r (lanes 16-31)
    #pragma unroll
    for (int i = 0; i < 4; ++i) {
        #pragma unroll
        for (int j = 0; j < 4; ++j) {
            const int n = nBase + wn * 64 + j * 16 + laneMod;
            const float bn = bias[n];
            #pragma unroll
            for (int r = 0; r < 8; ++r) {
                const int m = mBase + wm * 64 + i * 16 + r + 8 * laneHalf;
                float v = acc[i][j][r] + bn;
                if (RELU) v = v > 0.f ? v : 0.f;
                const size_t off = (size_t)m * N + n;
                if (OUT_BF16) ((__hip_bfloat16*)C)[off] = __float2bfloat16(v);
                else          ((float*)C)[off] = v;
            }
        }
    }
}

// ---------------------------------------------------------------------------
// K3: gate logits + softmax.  One wave per batch row; lanes 0..23 = (g,e).
// ---------------------------------------------------------------------------
__global__ __launch_bounds__(256) void gates_kernel(
    const __hip_bfloat16* __restrict__ concat, // [B][FEAT]
    const float* __restrict__ gate_w,          // [3][FEAT][8]
    const float* __restrict__ gate_b,          // [3][8]
    float* __restrict__ gates)                 // [B][3][8] post-softmax
{
    const int wave = threadIdx.x >> 5;
    const int lane = threadIdx.x & 31;
    const int b = blockIdx.x * 8 + wave;
    const int g = lane >> 3, e = lane & 7;

    float accum = 0.f;
    if (lane < 24) {
        const __hip_bfloat16* crow = concat + (size_t)b * FEAT_;
        const float* wcol = gate_w + ((size_t)g * FEAT_) * NEXP_ + e;
        #pragma unroll 4
        for (int f = 0; f < FEAT_; ++f)
            accum += __bfloat162float(crow[f]) * wcol[(size_t)f * NEXP_];
        accum += gate_b[g * NEXP_ + e];
    }
    if (lane < 24) {
        float mx = accum;
        #pragma unroll
        for (int m = 4; m >= 1; m >>= 1) mx = fmaxf(mx, __shfl_xor(mx, m, 32));
        float p = __expf(accum - mx);
        float s = p;
        #pragma unroll
        for (int m = 4; m >= 1; m >>= 1) s += __shfl_xor(s, m, 32);
        gates[(size_t)b * (NG_ * NEXP_) + lane] = p / s;
    }
}

// ---------------------------------------------------------------------------
// K4: mixed[g][b][h] = sum_e gates[b][g][e] * expert_out[b][e*1024+h]  (bf16)
// ---------------------------------------------------------------------------
__global__ __launch_bounds__(256) void mix_kernel(
    const __hip_bfloat16* __restrict__ eout,  // [B][8192]
    const float* __restrict__ gates,          // [B][3][8]
    __hip_bfloat16* __restrict__ mixed)       // [3][B][1024]
{
    size_t gid = (size_t)blockIdx.x * blockDim.x + threadIdx.x;
    int b = (int)(gid / ESZ_);
    int h = (int)(gid % ESZ_);
    const float* gb = gates + (size_t)b * (NG_ * NEXP_);
    float a0 = 0.f, a1 = 0.f, a2 = 0.f;
    #pragma unroll
    for (int e = 0; e < NEXP_; ++e) {
        float v = __bfloat162float(eout[(size_t)b * NTOT_ + e * ESZ_ + h]);
        a0 += gb[0 * NEXP_ + e] * v;
        a1 += gb[1 * NEXP_ + e] * v;
        a2 += gb[2 * NEXP_ + e] * v;
    }
    mixed[(size_t)0 * BATCH_ * ESZ_ + (size_t)b * ESZ_ + h] = __float2bfloat16(a0);
    mixed[(size_t)1 * BATCH_ * ESZ_ + (size_t)b * ESZ_ + h] = __float2bfloat16(a1);
    mixed[(size_t)2 * BATCH_ * ESZ_ + (size_t)b * ESZ_ + h] = __float2bfloat16(a2);
}

// ---------------------------------------------------------------------------
// K5: head. One wave per batch row: 3x(512->2) dots, 2-way softmax, clip,
// scatter 7 outputs into flat d_out.
// ---------------------------------------------------------------------------
__global__ __launch_bounds__(256) void head_kernel(
    const float* __restrict__ tower,   // [3][B][512] (post-relu)
    const float* __restrict__ out_w,   // [3][512][2]
    const float* __restrict__ out_b,   // [3][2]
    float* __restrict__ out)           // flat outputs
{
    const int wave = threadIdx.x >> 5;
    const int lane = threadIdx.x & 31;
    const int b = blockIdx.x * 8 + wave;

    float a[NG_][2];
    #pragma unroll
    for (int g = 0; g < NG_; ++g) { a[g][0] = 0.f; a[g][1] = 0.f; }

    for (int t = lane; t < TSZ_; t += 32) {
        #pragma unroll
        for (int g = 0; g < NG_; ++g) {
            float tv = tower[(size_t)g * BATCH_ * TSZ_ + (size_t)b * TSZ_ + t];
            a[g][0] += tv * out_w[((size_t)g * TSZ_ + t) * 2 + 0];
            a[g][1] += tv * out_w[((size_t)g * TSZ_ + t) * 2 + 1];
        }
    }
    #pragma unroll
    for (int g = 0; g < NG_; ++g)
        #pragma unroll
        for (int o = 0; o < 2; ++o)
            #pragma unroll
            for (int m = 16; m >= 1; m >>= 1)
                a[g][o] += __shfl_xor(a[g][o], m, 32);

    if (lane == 0) {
        float p0c[NG_], p1c[NG_];
        const float EPS = 1e-15f, HI = 1.0f - 1e-15f;
        #pragma unroll
        for (int g = 0; g < NG_; ++g) {
            float l0 = a[g][0] + out_b[g * 2 + 0];
            float l1 = a[g][1] + out_b[g * 2 + 1];
            float mx = fmaxf(l0, l1);
            float e0 = __expf(l0 - mx), e1 = __expf(l1 - mx);
            float inv = 1.0f / (e0 + e1);
            p0c[g] = fminf(fmaxf(e0 * inv, EPS), HI);
            p1c[g] = fminf(fmaxf(e1 * inv, EPS), HI);
        }
        const size_t B = BATCH_;
        out[2 * b + 0] = p0c[0];            // ctr_out [B,2]
        out[2 * b + 1] = p1c[0];
        out[2 * B + b] = p1c[0];            // ctr_prop_one [B,1]
        out[3 * B + 2 * b + 0] = p0c[1];    // cvr_out [B,2]
        out[3 * B + 2 * b + 1] = p1c[1];
        out[5 * B + b] = p1c[1];            // cvr_prop_one [B,1]
        float pc = p1c[0] * p1c[1];
        out[6 * B + 2 * b + 0] = 1.0f - pc; // ctcvr_prop [B,2]
        out[6 * B + 2 * b + 1] = pc;
        out[8 * B + b] = pc;                // ctcvr_prop_one [B,1]
        out[9 * B + b] = p1c[2];            // imp_prop_one [B,1]
    }
}

// ---------------------------------------------------------------------------
// Workspace layout (bytes)
// ---------------------------------------------------------------------------
constexpr size_t O_CONCAT = 0;                                    // bf16 [B][FEAT]
constexpr size_t SZ_CONCAT = (size_t)BATCH_ * FEAT_ * 2;
constexpr size_t O_WQ     = O_CONCAT + SZ_CONCAT;                 // bf16 [FEAT][8192]
constexpr size_t SZ_WQ     = (size_t)FEAT_ * NTOT_ * 2;
constexpr size_t O_EOUT   = O_WQ + SZ_WQ;                         // bf16 [B][8192]
constexpr size_t SZ_EOUT   = (size_t)BATCH_ * NTOT_ * 2;
constexpr size_t O_GATES  = O_EOUT + SZ_EOUT;                     // f32 [B][24]
constexpr size_t SZ_GATES  = (size_t)BATCH_ * NG_ * NEXP_ * 4;
constexpr size_t O_MIXED  = O_GATES + SZ_GATES;                   // bf16 [3][B][1024]
constexpr size_t SZ_MIXED  = (size_t)NG_ * BATCH_ * ESZ_ * 2;
constexpr size_t O_TWQ    = O_MIXED + SZ_MIXED;                   // bf16 [3][1024][512]
constexpr size_t SZ_TWQ    = (size_t)NG_ * ESZ_ * TSZ_ * 2;
constexpr size_t O_TOWER  = O_TWQ + SZ_TWQ;                       // f32 [3][B][512]

extern "C" void kernel_launch(void* const* d_in, const int* in_sizes, int n_in,
                              void* d_out, int out_size, void* d_ws, size_t ws_size,
                              hipStream_t stream) {
    const int*   inputs   = (const int*)  d_in[0];
    const float* emb      = (const float*)d_in[1];
    const float* expert_w = (const float*)d_in[2];
    const float* expert_b = (const float*)d_in[3];
    const float* gate_w   = (const float*)d_in[4];
    const float* gate_b   = (const float*)d_in[5];
    const float* tower_w  = (const float*)d_in[6];
    const float* tower_b  = (const float*)d_in[7];
    const float* out_w    = (const float*)d_in[8];
    const float* out_b    = (const float*)d_in[9];
    float* out = (float*)d_out;

    char* ws = (char*)d_ws;
    __hip_bfloat16* concatq = (__hip_bfloat16*)(ws + O_CONCAT);
    __hip_bfloat16* wq      = (__hip_bfloat16*)(ws + O_WQ);
    __hip_bfloat16* eoutq   = (__hip_bfloat16*)(ws + O_EOUT);
    float*          gates   = (float*)         (ws + O_GATES);
    __hip_bfloat16* mixedq  = (__hip_bfloat16*)(ws + O_MIXED);
    __hip_bfloat16* twq     = (__hip_bfloat16*)(ws + O_TWQ);
    float*          tower   = (float*)         (ws + O_TOWER);

    // K0: weight repack/cast to bf16
    {
        size_t n = (size_t)NEXP_ * FEAT_ * ESZ_;
        repack_expert_w_kernel<<<dim3((unsigned)((n + 255) / 256)), 256, 0, stream>>>(expert_w, wq);
        int nt = NG_ * ESZ_ * TSZ_;
        cast_bf16_kernel<<<dim3((nt + 255) / 256), 256, 0, stream>>>(tower_w, twq, nt);
    }
    // K1: embedding pool -> bf16 concat
    embed_pool_kernel<<<dim3(BATCH_, NFLD_), 128, 0, stream>>>(inputs, emb, concatq);

    // K2: expert GEMM  (16384 x 2944) @ (2944 x 8192), relu, bf16 out
    gemm_bf16_kernel<true, true><<<dim3(NTOT_ / TN, BATCH_ / TM), 256, 0, stream>>>(
        concatq, wq, expert_b, (void*)eoutq, BATCH_, NTOT_, FEAT_);

    // K3: gates softmax
    gates_kernel<<<dim3(BATCH_ / 8), 256, 0, stream>>>(concatq, gate_w, gate_b, gates);

    // K4: gate-weighted expert mixing -> bf16 mixed [3][B][1024]
    mix_kernel<<<dim3((unsigned)(((size_t)BATCH_ * ESZ_) / 256)), 256, 0, stream>>>(
        eoutq, gates, mixedq);

    // K5: tower GEMMs  (16384 x 1024) @ (1024 x 512) x3, relu, f32 out
    for (int g = 0; g < NG_; ++g) {
        gemm_bf16_kernel<true, false><<<dim3(TSZ_ / TN, BATCH_ / TM), 256, 0, stream>>>(
            mixedq + (size_t)g * BATCH_ * ESZ_,
            twq    + (size_t)g * ESZ_ * TSZ_,
            tower_b + g * TSZ_,
            (void*)(tower + (size_t)g * BATCH_ * TSZ_),
            BATCH_, TSZ_, ESZ_);
    }

    // K6: head + output scatter
    head_kernel<<<dim3(BATCH_ / 8), 256, 0, stream>>>(tower, out_w, out_b, out);
}
